// VGCNEncoder_64750926954679
// MI455X (gfx1250) — matrix operand order repaired and checked by reading, not compiled, not run on hardware
//
#include <hip/hip_runtime.h>
#include <hip/hip_bf16.h>

typedef __attribute__((ext_vector_type(16))) _Float16 v16h;
typedef __attribute__((ext_vector_type(8)))  _Float16 v8h;
typedef __attribute__((ext_vector_type(8)))  float    v8f;

#define NN 100000
#define NE 1600000
#define DIM 128
#define ODIM 64

// ---------------- degree / dinv ----------------
__global__ void k_deg_init(float* deg) {
    int i = blockIdx.x * blockDim.x + threadIdx.x;
    if (i < NN) deg[i] = 1.0f;   // self loop
}

__global__ void k_deg_count(const int* ei, float* deg) {
    int e = blockIdx.x * blockDim.x + threadIdx.x;
    if (e < NE) atomicAdd(&deg[ei[NE + e]], 1.0f);
}

__global__ void k_dinv(float* deg) {
    int i = blockIdx.x * blockDim.x + threadIdx.x;
    if (i < NN) deg[i] = rsqrtf(deg[i]);   // deg >= 1 always
}

// ---------------- row l2norm (128 cols) -> f16 ----------------
__global__ void k_l2norm_to_h16(const float* __restrict__ x, _Float16* __restrict__ out) {
    int wave = threadIdx.x >> 5, lane = threadIdx.x & 31;
    int row = blockIdx.x * 8 + wave;
    if (row >= NN) return;
    float4 v = reinterpret_cast<const float4*>(x + (size_t)row * DIM)[lane];
    float s = v.x * v.x + v.y * v.y + v.z * v.z + v.w * v.w;
    for (int m = 16; m >= 1; m >>= 1) s += __shfl_xor(s, m, 32);
    float inv = 1.0f / fmaxf(sqrtf(s), 1e-12f);
    _Float16* o = out + (size_t)row * DIM + lane * 4;
    o[0] = (_Float16)(v.x * inv);
    o[1] = (_Float16)(v.y * inv);
    o[2] = (_Float16)(v.z * inv);
    o[3] = (_Float16)(v.w * inv);
}

// ---------------- dual 64-col l2norm in place ----------------
__global__ void k_l2norm_dual(float* __restrict__ Y) {
    int wave = threadIdx.x >> 5, lane = threadIdx.x & 31;
    int row = blockIdx.x * 8 + wave;
    if (row >= NN) return;
    float4* p = reinterpret_cast<float4*>(Y + (size_t)row * DIM);
    float4 v = p[lane];
    float s = v.x * v.x + v.y * v.y + v.z * v.z + v.w * v.w;
    for (int m = 8; m >= 1; m >>= 1) s += __shfl_xor(s, m, 16);  // per 16-lane half (64 cols)
    float inv = 1.0f / fmaxf(sqrtf(s), 1e-12f);
    v.x *= inv; v.y *= inv; v.z *= inv; v.w *= inv;
    p[lane] = v;
}

// ---------------- weight packing into WMMA B-fragment layout ----------------
// Wp flat index t = ((ntile*4 + kchunk)*32 + lane)*16 + i ; element = W[k][n]
// with n = ntile*16 + (lane&15), k = kchunk*32 + (lane>>4)*16 + i
__global__ void k_pack_w0(const float* __restrict__ W, _Float16* __restrict__ Wp) {
    int t = blockIdx.x * 256 + threadIdx.x;
    if (t >= 8 * 4 * 32 * 16) return;
    int i = t & 15, lane = (t >> 4) & 31, kc = (t >> 9) & 3, nt = t >> 11;
    int n = nt * 16 + (lane & 15);
    int k = kc * 32 + (lane >> 4) * 16 + i;
    Wp[t] = (_Float16)W[(size_t)k * DIM + n];
}

__global__ void k_pack_wcat(const float* __restrict__ Wmu, const float* __restrict__ Wvar,
                            _Float16* __restrict__ Wp) {
    int t = blockIdx.x * 256 + threadIdx.x;
    if (t >= 8 * 4 * 32 * 16) return;
    int i = t & 15, lane = (t >> 4) & 31, kc = (t >> 9) & 3, nt = t >> 11;
    int n = nt * 16 + (lane & 15);
    int k = kc * 32 + (lane >> 4) * 16 + i;
    float v = (n < ODIM) ? Wmu[(size_t)k * ODIM + n] : Wvar[(size_t)k * ODIM + (n - ODIM)];
    Wp[t] = (_Float16)v;
}

__global__ void k_bcat(const float* bmu, const float* bvar, float* bcat) {
    int i = threadIdx.x;
    if (i < DIM) bcat[i] = (i < ODIM) ? bmu[i] : bvar[i - ODIM];
}

// ---------------- WMMA GEMM: [NN,128]f16 @ packed[128,128]f16 + bias -> [NN,128]f32 ----
// 1 block = 8 waves; wave w computes the 16x16 tile at (mbase, 16*w).
__global__ void k_gemm128(const _Float16* __restrict__ A, const _Float16* __restrict__ Wp,
                          const float* __restrict__ bias, float* __restrict__ Y) {
    int wave = threadIdx.x >> 5, lane = threadIdx.x & 31;
    int g = lane >> 4, r = lane & 15;
    int mbase = blockIdx.x * 16;
    const _Float16* Arow = A + (size_t)(mbase + r) * DIM;
    const v16h* Bp = reinterpret_cast<const v16h*>(Wp) + (size_t)(wave * 4) * 32 + lane;
    v8f acc = {};
#pragma unroll
    for (int kc = 0; kc < 4; ++kc) {
        // A fragment (contiguous k, matches packed B): element i = A[r][k0+16g+i]
        v16h a = *reinterpret_cast<const v16h*>(Arow + kc * 32 + 16 * g);
        v16h b = Bp[(size_t)kc * 32];
        acc = __builtin_amdgcn_wmma_f32_16x16x32_f16(false, a, false, b,
                                                     (short)0, acc, false, false);
    }
    // C/D layout: VGPR j -> M = g*8 + j, N = lane&15 (+ wave tile offset)
    int n = wave * 16 + r;
    float bv = bias[n];
    float* Yp = Y + (size_t)(mbase + g * 8) * DIM + n;
#pragma unroll
    for (int j = 0; j < 8; ++j) Yp[(size_t)j * DIM] = acc[j] + bv;
}

// ---------------- propagation over 128 cols into ws accumulator ----------------
__global__ void k_prop_self128(const float* __restrict__ X, const float* __restrict__ dinv,
                               float* __restrict__ acc) {
    int t = blockIdx.x * 256 + threadIdx.x;           // NN*32 threads
    if (t >= NN * 32) return;
    int i = t >> 5, lane = t & 31;
    float w = dinv[i]; w *= w;
    float4 v = reinterpret_cast<const float4*>(X + (size_t)i * DIM)[lane];
    float4 o; o.x = v.x * w; o.y = v.y * w; o.z = v.z * w; o.w = v.w * w;
    reinterpret_cast<float4*>(acc + (size_t)i * DIM)[lane] = o;
}

__global__ void k_prop_edge128(const int* __restrict__ ei, const float* __restrict__ X,
                               const float* __restrict__ dinv, float* __restrict__ acc) {
    int wave = threadIdx.x >> 5, lane = threadIdx.x & 31;
    int e = blockIdx.x * 8 + wave;
    if (e >= NE) return;
    int s = ei[e], d = ei[NE + e];
    float w = dinv[s] * dinv[d];
    float4 v = reinterpret_cast<const float4*>(X + (size_t)s * DIM)[lane];
    float* dst = acc + (size_t)d * DIM + lane * 4;
    atomicAdd(dst + 0, v.x * w);
    atomicAdd(dst + 1, v.y * w);
    atomicAdd(dst + 2, v.z * w);
    atomicAdd(dst + 3, v.w * w);
}

__global__ void k_relu_h16(const float* __restrict__ acc, _Float16* __restrict__ out) {
    size_t t = (size_t)blockIdx.x * 256 + threadIdx.x;   // NN*128
    if (t >= (size_t)NN * DIM) return;
    out[t] = (_Float16)fmaxf(acc[t], 0.0f);
}

// ---------------- final propagation straight into d_out (mu || var split) ----------
__device__ __forceinline__ float* out_ptr(float* out, int i, int c) {
    return (c < ODIM) ? (out + (size_t)i * ODIM + c)
                      : (out + (size_t)NN * ODIM + (size_t)i * ODIM + (c - ODIM));
}

__global__ void k_prop_self_out(const float* __restrict__ X, const float* __restrict__ dinv,
                                float* __restrict__ out) {
    int t = blockIdx.x * 256 + threadIdx.x;           // NN*32 threads
    if (t >= NN * 32) return;
    int i = t >> 5, c = (t & 31) * 4;                 // 4-col chunk never crosses the 64 split
    float w = dinv[i]; w *= w;
    float4 v = reinterpret_cast<const float4*>(X + (size_t)i * DIM)[t & 31];
    float4 o; o.x = v.x * w; o.y = v.y * w; o.z = v.z * w; o.w = v.w * w;
    *reinterpret_cast<float4*>(out_ptr(out, i, c)) = o;
}

__global__ void k_prop_edge_out(const int* __restrict__ ei, const float* __restrict__ X,
                                const float* __restrict__ dinv, float* __restrict__ out) {
    int wave = threadIdx.x >> 5, lane = threadIdx.x & 31;
    int e = blockIdx.x * 8 + wave;
    if (e >= NE) return;
    int s = ei[e], d = ei[NE + e];
    float w = dinv[s] * dinv[d];
    int c = lane * 4;
    float4 v = reinterpret_cast<const float4*>(X + (size_t)s * DIM)[lane];
    float* dst = out_ptr(out, d, c);
    atomicAdd(dst + 0, v.x * w);
    atomicAdd(dst + 1, v.y * w);
    atomicAdd(dst + 2, v.z * w);
    atomicAdd(dst + 3, v.w * w);
}

extern "C" void kernel_launch(void* const* d_in, const int* in_sizes, int n_in,
                              void* d_out, int out_size, void* d_ws, size_t ws_size,
                              hipStream_t stream) {
    const float* x     = (const float*)d_in[0];
    const int*   ei    = (const int*)d_in[1];
    const float* W0    = (const float*)d_in[2];
    const float* b0    = (const float*)d_in[3];
    const float* Wmu   = (const float*)d_in[4];
    const float* bmu   = (const float*)d_in[5];
    const float* Wvar  = (const float*)d_in[6];
    const float* bvar  = (const float*)d_in[7];
    float* out = (float*)d_out;

    // workspace carve-up (256B aligned)
    char* ws = (char*)d_ws;
    size_t off = 0;
    auto take = [&](size_t bytes) { char* p = ws + off; off = (off + bytes + 255) & ~(size_t)255; return p; };
    float*    dinv = (float*)take((size_t)NN * 4);
    _Float16* hh   = (_Float16*)take((size_t)NN * DIM * 2);   // f16 activations (reused)
    float*    z0   = (float*)take((size_t)NN * DIM * 4);      // GEMM out (reused as y2)
    float*    acc  = (float*)take((size_t)NN * DIM * 4);      // propagation accumulator
    _Float16* W0p  = (_Float16*)take(8 * 4 * 32 * 16 * 2);
    _Float16* Wcp  = (_Float16*)take(8 * 4 * 32 * 16 * 2);
    float*    bcat = (float*)take(DIM * 4);
    (void)ws_size; (void)n_in; (void)in_sizes; (void)out_size;

    const int T = 256;
    // degrees -> dinv (shared by all three propagations)
    k_deg_init <<<(NN + T - 1) / T, T, 0, stream>>>(dinv);
    k_deg_count<<<(NE + T - 1) / T, T, 0, stream>>>(ei, dinv);
    k_dinv     <<<(NN + T - 1) / T, T, 0, stream>>>(dinv);

    // one-shot weight packs
    k_pack_w0  <<<64, T, 0, stream>>>(W0, W0p);
    k_pack_wcat<<<64, T, 0, stream>>>(Wmu, Wvar, Wcp);
    k_bcat     <<<1, 128, 0, stream>>>(bmu, bvar, bcat);

    // h = l2norm(x) -> f16
    k_l2norm_to_h16<<<NN / 8, T, 0, stream>>>(x, hh);
    // z0 = h @ W0 + b0   (WMMA)
    k_gemm128<<<NN / 16, T, 0, stream>>>(hh, W0p, b0, z0);
    // h1 = relu(propagate(z0))
    k_prop_self128<<<(NN * 32) / T, T, 0, stream>>>(z0, dinv, acc);
    k_prop_edge128<<<NE / 8, T, 0, stream>>>(ei, z0, dinv, acc);
    k_relu_h16<<<(NN * DIM) / T, T, 0, stream>>>(acc, hh);
    // y = h1 @ [W_mu | W_var] + [b_mu | b_var]   (WMMA, fused heads)
    k_gemm128<<<NN / 16, T, 0, stream>>>(hh, Wcp, bcat, z0);
    // per-head 64-col l2norm, in place
    k_l2norm_dual<<<NN / 8, T, 0, stream>>>(z0);
    // mu/var = propagate(y), written directly into d_out with the split mapping
    k_prop_self_out<<<(NN * 32) / T, T, 0, stream>>>(z0, dinv, out);
    k_prop_edge_out<<<NE / 8, T, 0, stream>>>(ei, z0, dinv, out);
}
